// ScaledDotProductAttention_38336878084625
// MI455X (gfx1250) — matrix operand order repaired and checked by reading, not compiled, not run on hardware
//
#include <hip/hip_runtime.h>
#include <stdint.h>

typedef __attribute__((ext_vector_type(16))) _Float16     v16h;
typedef __attribute__((ext_vector_type(8)))  float        v8f;
typedef __attribute__((ext_vector_type(4)))  unsigned int v4u;
typedef __attribute__((ext_vector_type(8)))  int          v8i;
typedef __attribute__((ext_vector_type(4)))  int          v4i;

constexpr int Bc = 4, Hc = 16, Sc = 2048, Dc = 64;
constexpr int WAVES = 8, QT = 16, KT = 32;
constexpr int QBLK = WAVES * QT;          // 128 q rows per workgroup
constexpr int NQB  = Sc / QBLK;           // 16 q-blocks per (b,h)
constexpr int NCH  = Sc / KT;             // 64 kv chunks
constexpr float SCALEF = 0.125f;          // 1/sqrt(64)
constexpr float NEGV   = -1.0e9f;
constexpr float LOG2E  = 1.4426950408889634f;
// LDS strides (in halfs), padded to spread banks
constexpr int KS = 72;                    // K chunk row stride (64 + pad)
constexpr int VS = 40;                    // V^T row stride (32 + pad)
constexpr int PS = 40;                    // P scratch row stride (32 + pad)

union U16x8 { uint32_t u[8]; v16h v; };

__device__ __forceinline__ float rmax16(float x) {
#pragma unroll
  for (int off = 8; off; off >>= 1) x = fmaxf(x, __shfl_xor(x, off, 16));
  return x;
}
__device__ __forceinline__ float rsum16(float x) {
#pragma unroll
  for (int off = 8; off; off >>= 1) x += __shfl_xor(x, off, 16);
  return x;
}

// Tensor Data Mover: load a 32(x) by 128(y) tile of i32 from a 2048x2048 i32
// tensor (row stride 2048) into LDS at lds_off. D# per CDNA5 ISA section 8.
// This toolchain exposes the 6-arg builtin: (g0, g1, g2, g3, g_extra, cpol).
__device__ __forceinline__ void tdm_mask_load(uint32_t lds_off, uint64_t ga) {
  v4u g0;
  g0[0] = 1u;                                            // count=1, user desc
  g0[1] = lds_off;                                       // lds_addr (bytes)
  g0[2] = (uint32_t)(ga & 0xFFFFFFFFu);                  // global_addr[31:0]
  g0[3] = (uint32_t)((ga >> 32) & 0x01FFFFFFu)           // global_addr[56:32]
        | 0x80000000u;                                   // type=2 ("image")
  v8i g1;
  g1[0] = 0x00020000;                                    // data_size=2 (4B)
  g1[1] = (int)((2048u & 0xFFFFu) << 16);                // tensor_dim0 lo16
  g1[2] = (int)((2048u & 0xFFFFu) << 16);                // dim0 hi | tensor_dim1 lo16
  g1[3] = (int)(32u << 16);                              // dim1 hi | tile_dim0=32
  g1[4] = 128;                                           // tile_dim1=128, tile_dim2=0
  g1[5] = 2048;                                          // tensor_dim0_stride lo32
  g1[6] = 0;                                             // stride hi | dim1_stride lo
  g1[7] = 0;
  v4i z4 = {0, 0, 0, 0};
  v8i z8 = {0, 0, 0, 0, 0, 0, 0, 0};
  __builtin_amdgcn_tensor_load_to_lds(g0, g1, z4, z4, z8, 0);
}

__global__ __launch_bounds__(256) void fa_fwd_f16wmma(
    const float* __restrict__ Q, const float* __restrict__ K,
    const float* __restrict__ V, const int* __restrict__ M,
    float* __restrict__ O) {
  __shared__ _Float16 Ksh[KT * KS];            // K chunk, f16 row-major
  __shared__ _Float16 Vt [Dc * VS];            // V chunk, f16 transposed [d][kv]
  __shared__ _Float16 Psh[WAVES * 16 * PS];    // per-wave P scratch
  __shared__ int      Msh[2][QBLK * KT];       // double-buffered mask tile (TDM dest)

  const int bh = blockIdx.x / NQB;
  const int qb = blockIdx.x % NQB;
  const int b  = bh / Hc;
  const size_t base = (size_t)bh * Sc * Dc;

  const int tid  = threadIdx.x;
  const int wave = tid >> 5;
  const int lane = tid & 31;
  const int hh   = lane >> 4;   // half-wave select (ISA 16-bit layouts)
  const int ln   = lane & 15;
  const int q0   = qb * QBLK + wave * QT;

  // ---- Q A-fragments (16x32 f16 A layout), d in [0,32) and [32,64)
  v16h qa0, qa1;
  {
    const float* Qp = Q + base + (size_t)(q0 + ln) * Dc;
#pragma unroll
    for (int i = 0; i < 8; i++) {
      const int kk = ((i < 4) ? (i * 2) : (16 + (i - 4) * 2)) + hh * 8;
      float2 a = *(const float2*)(Qp + kk);
      float2 c = *(const float2*)(Qp + 32 + kk);
      qa0[2 * i] = (_Float16)a.x; qa0[2 * i + 1] = (_Float16)a.y;
      qa1[2 * i] = (_Float16)c.x; qa1[2 * i + 1] = (_Float16)c.y;
    }
  }

  v8f o0 = {}, o1 = {}, o2 = {}, o3 = {};
  float mrun[8], lrun[8];
#pragma unroll
  for (int r = 0; r < 8; r++) { mrun[r] = -INFINITY; lrun[r] = 0.f; }

  // byte address of mask[b, 0, qb*128, 0]
  const uint64_t mga_base =
      (uint64_t)(uintptr_t)(M + (size_t)b * Sc * Sc + (size_t)(qb * QBLK) * Sc);

  // ---- register-pipelined K/V global loads (2 float4 each per thread)
  auto gload = [&](int c, float4* kr, float4* vr) {
#pragma unroll
    for (int it = 0; it < 2; it++) {
      const int fi  = tid + it * 256;
      const int row = fi >> 4;
      const int c4  = (fi & 15) << 2;
      kr[it] = *(const float4*)(K + base + (size_t)(c * KT + row) * Dc + c4);
      vr[it] = *(const float4*)(V + base + (size_t)(c * KT + row) * Dc + c4);
    }
  };

  float4 kcur[2], vcur[2], knxt[2], vnxt[2];
  gload(0, kcur, vcur);
  if (wave == 0) tdm_mask_load((uint32_t)(uintptr_t)&Msh[0][0], mga_base);

  auto body = [&](int c, float4* kc, float4* vc, float4* kn, float4* vn) {
    __syncthreads();   // (A) previous chunk's LDS (K/V/mask bufs) no longer in use

    // ---- stage chunk c from registers into LDS (K row-major, V transposed)
#pragma unroll
    for (int it = 0; it < 2; it++) {
      const int fi  = tid + it * 256;
      const int row = fi >> 4;
      const int c4  = (fi & 15) << 2;
      union { _Float16 h[4]; uint2 u; } pk;
      pk.h[0] = (_Float16)kc[it].x; pk.h[1] = (_Float16)kc[it].y;
      pk.h[2] = (_Float16)kc[it].z; pk.h[3] = (_Float16)kc[it].w;
      *(uint2*)&Ksh[row * KS + c4] = pk.u;
      Vt[(c4 + 0) * VS + row] = (_Float16)vc[it].x;
      Vt[(c4 + 1) * VS + row] = (_Float16)vc[it].y;
      Vt[(c4 + 2) * VS + row] = (_Float16)vc[it].z;
      Vt[(c4 + 3) * VS + row] = (_Float16)vc[it].w;
    }
    // issue next chunk's K/V loads; waits land one compute phase later
    if (c + 1 < NCH) gload(c + 1, kn, vn);
    // TDM: issue mask tile for c+1 into the other buffer, ensure tile c landed
    if (wave == 0) {
      if (c + 1 < NCH) {
        tdm_mask_load((uint32_t)(uintptr_t)&Msh[(c + 1) & 1][0],
                      mga_base + (uint64_t)(c + 1) * KT * sizeof(int));
        __builtin_amdgcn_s_wait_tensorcnt(1);
      } else {
        __builtin_amdgcn_s_wait_tensorcnt(0);
      }
    }
    __syncthreads();   // (B) K/V staged + mask tile c visible to all waves

    // ---- S = Q @ K^T : two 16x16 tiles (kv cols 0-15 and 16-31)
    v8f s0 = {}, s1 = {};
#pragma unroll
    for (int dch = 0; dch < 2; dch++) {
      U16x8 kb0, kb1;
      const _Float16* p0 = &Ksh[(ln     ) * KS + dch * 32 + hh * 8];
      const _Float16* p1 = &Ksh[(16 + ln) * KS + dch * 32 + hh * 8];
#pragma unroll
      for (int j = 0; j < 8; j++) {
        kb0.u[j] = *(const uint32_t*)(p0 + 2 * j + ((j < 4) ? 0 : 8));
        kb1.u[j] = *(const uint32_t*)(p1 + 2 * j + ((j < 4) ? 0 : 8));
      }
      const v16h qa = dch ? qa1 : qa0;
      s0 = __builtin_amdgcn_wmma_f32_16x16x32_f16(false, qa, false, kb0.v,
                                                  (short)0, s0, false, false);
      s1 = __builtin_amdgcn_wmma_f32_16x16x32_f16(false, qa, false, kb1.v,
                                                  (short)0, s1, false, false);
    }

    // ---- scale, mask (from TDM-staged LDS tile), online softmax
    const int* mtile = &Msh[c & 1][0];
#pragma unroll
    for (int r = 0; r < 8; r++) {
      const int* mr = mtile + (wave * 16 + r + 8 * hh) * KT + ln;
      float a0 = s0[r] * SCALEF; if (mr[0]  == 0) a0 = NEGV;
      float a1 = s1[r] * SCALEF; if (mr[16] == 0) a1 = NEGV;
      const float mx = rmax16(fmaxf(a0, a1));
      const float nm = fmaxf(mrun[r], mx);
      const float alpha = __builtin_amdgcn_exp2f((mrun[r] - nm) * LOG2E);
      const float p0 = __builtin_amdgcn_exp2f((a0 - nm) * LOG2E);
      const float p1 = __builtin_amdgcn_exp2f((a1 - nm) * LOG2E);
      lrun[r] = lrun[r] * alpha + rsum16(p0 + p1);
      mrun[r] = nm;
      o0[r] *= alpha; o1[r] *= alpha; o2[r] *= alpha; o3[r] *= alpha;
      const int prow = r + 8 * hh;
      Psh[wave * 16 * PS + prow * PS + ln]      = (_Float16)p0;
      Psh[wave * 16 * PS + prow * PS + 16 + ln] = (_Float16)p1;
    }

    // P scratch is per-wave; LDS ops are in-order within a wave, but make the
    // store->load ordering explicit before re-reading in A layout.
    asm volatile("s_wait_dscnt 0" ::: "memory");

    // ---- P as A-fragment (16x32)
    U16x8 pa;
    {
      const _Float16* Pp = &Psh[wave * 16 * PS + ln * PS];
#pragma unroll
      for (int i = 0; i < 8; i++) {
        const int kk = ((i < 4) ? (i * 2) : (16 + (i - 4) * 2)) + hh * 8;
        pa.u[i] = *(const uint32_t*)(Pp + kk);
      }
    }

    // ---- O += P @ V  (four 16-wide D chunks, K=32)
#pragma unroll
    for (int t = 0; t < 4; t++) {
      U16x8 vb;
      const _Float16* vp = &Vt[(t * 16 + ln) * VS + hh * 8];
#pragma unroll
      for (int j = 0; j < 8; j++) vb.u[j] = *(const uint32_t*)(vp + 2 * j + ((j < 4) ? 0 : 8));
      v8f& ot = (t == 0) ? o0 : (t == 1) ? o1 : (t == 2) ? o2 : o3;
      ot = __builtin_amdgcn_wmma_f32_16x16x32_f16(false, pa.v, false, vb.v,
                                                  (short)0, ot, false, false);
    }
  };

  for (int c = 0; c < NCH; c += 2) {   // NCH is even; buffers ping-pong
    body(c,     kcur, vcur, knxt, vnxt);
    body(c + 1, knxt, vnxt, kcur, vcur);
  }

  // ---- epilogue: O / l
#pragma unroll
  for (int r = 0; r < 8; r++) {
    const int row = q0 + r + 8 * hh;
    const float inv = 1.0f / lrun[r];
    float* op = O + base + (size_t)row * Dc + ln;
    op[0]  = o0[r] * inv;
    op[16] = o1[r] * inv;
    op[32] = o2[r] * inv;
    op[48] = o3[r] * inv;
  }
}

extern "C" void kernel_launch(void* const* d_in, const int* in_sizes, int n_in,
                              void* d_out, int out_size, void* d_ws, size_t ws_size,
                              hipStream_t stream) {
  const float* q = (const float*)d_in[0];
  const float* k = (const float*)d_in[1];
  const float* v = (const float*)d_in[2];
  const int*   m = (const int*)d_in[3];
  float* out = (float*)d_out;
  (void)in_sizes; (void)n_in; (void)out_size; (void)d_ws; (void)ws_size;
  dim3 grid(Bc * Hc * NQB);   // 1024 workgroups
  dim3 block(256);            // 8 wave32
  hipLaunchKernelGGL(fa_fwd_f16wmma, grid, block, 0, stream, q, k, v, m, out);
}
